// MFNA_PM_35356170780800
// MI455X (gfx1250) — matrix operand order repaired and checked by reading, not compiled, not run on hardware
//
#include <hip/hip_runtime.h>
#include <hip/hip_bf16.h>

typedef __attribute__((ext_vector_type(16))) __bf16 v16bf;
typedef __attribute__((ext_vector_type(8)))  float  v8f;

#define C_DIM 64
#define N_DIM 65536
#define B_DIM 4
#define NCHUNK 1024     /* N / 64 */
#define NT 64           /* columns per chunk */
#define LP 66           /* padded LDS row stride (u16) */
#define BLK3 256

union Frag { unsigned int u[8]; v16bf b; };

__device__ __forceinline__ unsigned short f2bf(float x) {
  unsigned int u = __builtin_bit_cast(unsigned int, x);
  u += 0x7FFFu + ((u >> 16) & 1u);            // round-to-nearest-even
  return (unsigned short)(u >> 16);
}
__device__ __forceinline__ float bf2f(unsigned short h) {
  unsigned int u = ((unsigned int)h) << 16;
  return __builtin_bit_cast(float, u);
}
__device__ __forceinline__ unsigned int pack2(float a, float b) {
  return (unsigned int)f2bf(a) | ((unsigned int)f2bf(b) << 16);
}
__device__ __forceinline__ unsigned int ldsD(const unsigned short* s, int i) {
  return *(const unsigned int*)(s + i);       // i always even -> 4B aligned
}
__device__ __forceinline__ float softplusf(float x) {
  return (x > 20.0f) ? x : log1pf(__expf(x));
}

// Stage a 64x64 f32 tile (global, row stride N_DIM) into LDS as bf16 [n][c],
// using b128 global loads (4 floats along n per thread).
__device__ __forceinline__ void stage_tile(const float* __restrict__ xbase, int n0,
                                           unsigned short* __restrict__ lds, int tid) {
  const float4* xb4 = (const float4*)(xbase + n0);
  #pragma unroll
  for (int e = tid; e < C_DIM * (NT / 4); e += 256) {   // 1024 float4s, 4 iters
    int c = e >> 4, n4 = e & 15;
    float4 x = xb4[(size_t)c * (N_DIM / 4) + n4];
    int n = n4 * 4;
    lds[(n + 0) * LP + c] = f2bf(x.x);
    lds[(n + 1) * LP + c] = f2bf(x.y);
    lds[(n + 2) * LP + c] = f2bf(x.z);
    lds[(n + 3) * LP + c] = f2bf(x.w);
  }
}

// ---------------- Kernel 1: k/v projection + KV reduction partials ----------------
__global__ __launch_bounds__(256) void k1_proj_kv(
    const float* __restrict__ low,
    const float* __restrict__ Wk, const float* __restrict__ bk,
    const float* __restrict__ Wv, const float* __restrict__ bv,
    float* __restrict__ attnP, float* __restrict__ ksumP, int P)
{
  __shared__ __attribute__((aligned(16))) unsigned short X_lds[NT * LP];    // [n][c] bf16
  __shared__ __attribute__((aligned(16))) unsigned short K_lds[C_DIM * LP]; // [m][n] bf16
  __shared__ __attribute__((aligned(16))) unsigned short V_lds[C_DIM * LP]; // [c][n] bf16
  __shared__ float ksum_s[C_DIM];

  const int tid = threadIdx.x;
  const int w    = tid >> 5;
  const int lane = tid & 31;
  const int hi   = lane >> 4;
  const int l15  = lane & 15;
  const int b   = blockIdx.x / P;
  const int blk = blockIdx.x % P;

  const bool  isK    = (w < 4);
  const int   m_base = (w & 3) * 16;
  const float* Wsrc  = isK ? Wk : Wv;
  const float* bsrc  = isK ? bk : bv;

  // Preload weight A-fragments (M = out channel rows, K = in channels, 2 frags for K=64)
  Frag fragW[2];
  {
    const int row = m_base + l15;
    #pragma unroll
    for (int kk = 0; kk < 2; ++kk)
      #pragma unroll
      for (int j = 0; j < 8; ++j) {
        int p = j + hi * 8 + kk * 16;
        fragW[kk].u[j] = pack2(Wsrc[row * C_DIM + 2 * p], Wsrc[row * C_DIM + 2 * p + 1]);
      }
  }
  float biasv[8];
  #pragma unroll
  for (int r = 0; r < 8; ++r) biasv[r] = bsrc[m_base + r + hi * 8];

  // Persistent attention accumulators: wave owns tiles (m_tA, cA0) and (m_tA, cA0+1)
  const int m_tA = w >> 1;
  const int cA0  = 2 * (w & 1);
  v8f accA0, accA1;
  #pragma unroll
  for (int r = 0; r < 8; ++r) { accA0[r] = 0.0f; accA1[r] = 0.0f; }

  if (tid < C_DIM) ksum_s[tid] = 0.0f;
  __syncthreads();

  const float* xbase = low + (size_t)b * C_DIM * N_DIM;

  for (int chunk = blk; chunk < NCHUNK; chunk += P) {
    const int n0 = chunk * NT;

    stage_tile(xbase, n0, X_lds, tid);
    __syncthreads();

    // Projection GEMM: this wave computes one 16-row band of k OR v, all 4 n-tiles
    unsigned short* dst = isK ? K_lds : V_lds;
    #pragma unroll
    for (int nt = 0; nt < 4; ++nt) {
      const int ncol = nt * 16 + l15;
      v8f acc;
      #pragma unroll
      for (int r = 0; r < 8; ++r) acc[r] = biasv[r];
      Frag b0, b1;
      #pragma unroll
      for (int j = 0; j < 8; ++j) {
        b0.u[j] = ldsD(X_lds, ncol * LP +      hi * 16 + 2 * j);
        b1.u[j] = ldsD(X_lds, ncol * LP + 32 + hi * 16 + 2 * j);
      }
      acc = __builtin_amdgcn_wmma_f32_16x16x32_bf16(false, fragW[0].b, false, b0.b, (short)0, acc, false, false);
      acc = __builtin_amdgcn_wmma_f32_16x16x32_bf16(false, fragW[1].b, false, b1.b, (short)0, acc, false, false);
      #pragma unroll
      for (int r = 0; r < 8; ++r) {
        float v = acc[r];
        if (isK) v = softplusf(v);
        dst[(m_base + r + hi * 8) * LP + ncol] = f2bf(v);
      }
    }
    __syncthreads();

    // ksum over this chunk (one thread per channel, dword reads, fixed order)
    if (tid < C_DIM) {
      float s = 0.0f;
      #pragma unroll 8
      for (int n2 = 0; n2 < NT / 2; ++n2) {
        unsigned int d = ldsD(K_lds, tid * LP + 2 * n2);
        s += bf2f((unsigned short)(d & 0xFFFFu)) + bf2f((unsigned short)(d >> 16));
      }
      ksum_s[tid] += s;
    }

    // attn += k_tile @ v_tile^T   (A = k rows, K = n;  B = v^T, col = c)
    Frag ak[2];
    #pragma unroll
    for (int kk = 0; kk < 2; ++kk)
      #pragma unroll
      for (int j = 0; j < 8; ++j) {
        int p = j + hi * 8;
        ak[kk].u[j] = ldsD(K_lds, (m_tA * 16 + l15) * LP + kk * 32 + 2 * p);
      }
    #pragma unroll
    for (int i = 0; i < 2; ++i) {
      const int crow = (cA0 + i) * 16 + l15;
      Frag q0, q1;
      #pragma unroll
      for (int j = 0; j < 8; ++j) {
        q0.u[j] = ldsD(V_lds, crow * LP +      hi * 16 + 2 * j);
        q1.u[j] = ldsD(V_lds, crow * LP + 32 + hi * 16 + 2 * j);
      }
      v8f& acc = i ? accA1 : accA0;
      acc = __builtin_amdgcn_wmma_f32_16x16x32_bf16(false, ak[0].b, false, q0.b, (short)0, acc, false, false);
      acc = __builtin_amdgcn_wmma_f32_16x16x32_bf16(false, ak[1].b, false, q1.b, (short)0, acc, false, false);
    }
    __syncthreads();
  }

  // Write per-block partials
  float* ap = attnP + (size_t)(b * P + blk) * (C_DIM * C_DIM);
  #pragma unroll
  for (int i = 0; i < 2; ++i) {
    const v8f& acc = i ? accA1 : accA0;
    const int ccol = (cA0 + i) * 16 + l15;
    #pragma unroll
    for (int r = 0; r < 8; ++r)
      ap[(m_tA * 16 + r + hi * 8) * C_DIM + ccol] = acc[r];
  }
  if (tid < C_DIM) ksumP[(size_t)(b * P + blk) * C_DIM + tid] = ksum_s[tid];
}

// ---------------- Kernel 2: reduce partials ----------------
__global__ __launch_bounds__(256) void k2_reduce(
    const float* __restrict__ attnP, const float* __restrict__ ksumP,
    float* __restrict__ attnF, float* __restrict__ ksumF, int P)
{
  const int b = blockIdx.x, tid = threadIdx.x;
  for (int idx = tid; idx < C_DIM * C_DIM; idx += 256) {
    float s = 0.0f;
    for (int p = 0; p < P; ++p) s += attnP[((size_t)(b * P + p)) * (C_DIM * C_DIM) + idx];
    attnF[(size_t)b * C_DIM * C_DIM + idx] = s;
  }
  if (tid < C_DIM) {
    float s = 1e-6f;  // fold EPS into ksum here
    for (int p = 0; p < P; ++p) s += ksumP[(size_t)(b * P + p) * C_DIM + tid];
    ksumF[b * C_DIM + tid] = s;
  }
}

// ---------------- Kernel 3: q projection + normalized output GEMM ----------------
__global__ __launch_bounds__(256) void k3_out(
    const float* __restrict__ high,
    const float* __restrict__ Wq, const float* __restrict__ bq,
    const float* __restrict__ attnF, const float* __restrict__ ksumF,
    float* __restrict__ out)
{
  __shared__ __attribute__((aligned(16))) unsigned short X_lds[NT * LP];    // [n][c]
  __shared__ __attribute__((aligned(16))) unsigned short Q_lds[NT * LP];    // [n][m]
  __shared__ float ksum_s[C_DIM];
  __shared__ float norm_s[NT];

  const int tid = threadIdx.x;
  const int w    = tid >> 5;
  const int lane = tid & 31;
  const int hi   = lane >> 4;
  const int l15  = lane & 15;
  const int b   = blockIdx.x / BLK3;
  const int blk = blockIdx.x % BLK3;

  const int mq_base = (w >> 1) * 16;  // q rows this wave computes
  const int cT      = w >> 1;         // output channel tile this wave computes
  const int ntb     = 2 * (w & 1);    // first of the wave's two n-tiles

  // Wq A-fragments
  Frag fragWq[2];
  {
    const int row = mq_base + l15;
    #pragma unroll
    for (int kk = 0; kk < 2; ++kk)
      #pragma unroll
      for (int j = 0; j < 8; ++j) {
        int p = j + hi * 8 + kk * 16;
        fragWq[kk].u[j] = pack2(Wq[row * C_DIM + 2 * p], Wq[row * C_DIM + 2 * p + 1]);
      }
  }
  // attn^T A-fragments: A[row=c][k=m] = attnF[m*64 + c]
  Frag fragAt[2];
  {
    const int crow = cT * 16 + l15;
    const float* A = attnF + (size_t)b * C_DIM * C_DIM;
    #pragma unroll
    for (int kk = 0; kk < 2; ++kk)
      #pragma unroll
      for (int j = 0; j < 8; ++j) {
        int p  = j + hi * 8;
        int m0 = kk * 32 + 2 * p;
        fragAt[kk].u[j] = pack2(A[m0 * C_DIM + crow], A[(m0 + 1) * C_DIM + crow]);
      }
  }
  float biasq[8];
  #pragma unroll
  for (int r = 0; r < 8; ++r) biasq[r] = bq[mq_base + r + hi * 8];
  if (tid < C_DIM) ksum_s[tid] = ksumF[b * C_DIM + tid];
  __syncthreads();

  const float* xbase = high + (size_t)b * C_DIM * N_DIM;
  float*       obase = out  + (size_t)b * C_DIM * N_DIM;

  for (int chunk = blk; chunk < NCHUNK; chunk += BLK3) {
    const int n0 = chunk * NT;

    stage_tile(xbase, n0, X_lds, tid);
    __syncthreads();

    // q = softplus(Wq x + bq), staged to LDS as [n][m] bf16
    #pragma unroll
    for (int i = 0; i < 2; ++i) {
      const int ncol = (ntb + i) * 16 + l15;
      v8f acc;
      #pragma unroll
      for (int r = 0; r < 8; ++r) acc[r] = biasq[r];
      Frag b0, b1;
      #pragma unroll
      for (int j = 0; j < 8; ++j) {
        b0.u[j] = ldsD(X_lds, ncol * LP +      hi * 16 + 2 * j);
        b1.u[j] = ldsD(X_lds, ncol * LP + 32 + hi * 16 + 2 * j);
      }
      acc = __builtin_amdgcn_wmma_f32_16x16x32_bf16(false, fragWq[0].b, false, b0.b, (short)0, acc, false, false);
      acc = __builtin_amdgcn_wmma_f32_16x16x32_bf16(false, fragWq[1].b, false, b1.b, (short)0, acc, false, false);
      #pragma unroll
      for (int r = 0; r < 8; ++r)
        Q_lds[ncol * LP + mq_base + r + hi * 8] = f2bf(softplusf(acc[r]));
    }
    __syncthreads();

    // norm[n] = 1 / (q[:,n] . (ksum + EPS)), dword reads over m
    if (tid < NT) {
      float d = 0.0f;
      #pragma unroll 8
      for (int m2 = 0; m2 < C_DIM / 2; ++m2) {
        unsigned int q2 = ldsD(Q_lds, tid * LP + 2 * m2);
        d += bf2f((unsigned short)(q2 & 0xFFFFu)) * ksum_s[2 * m2]
           + bf2f((unsigned short)(q2 >> 16))     * ksum_s[2 * m2 + 1];
      }
      norm_s[tid] = 1.0f / d;
    }
    __syncthreads();

    // out[:,n] = norm[n] * attn^T @ q[:,n]
    #pragma unroll
    for (int i = 0; i < 2; ++i) {
      const int ncol = (ntb + i) * 16 + l15;
      v8f acc;
      #pragma unroll
      for (int r = 0; r < 8; ++r) acc[r] = 0.0f;
      Frag b0, b1;
      #pragma unroll
      for (int j = 0; j < 8; ++j) {
        b0.u[j] = ldsD(Q_lds, ncol * LP +      hi * 16 + 2 * j);
        b1.u[j] = ldsD(Q_lds, ncol * LP + 32 + hi * 16 + 2 * j);
      }
      acc = __builtin_amdgcn_wmma_f32_16x16x32_bf16(false, fragAt[0].b, false, b0.b, (short)0, acc, false, false);
      acc = __builtin_amdgcn_wmma_f32_16x16x32_bf16(false, fragAt[1].b, false, b1.b, (short)0, acc, false, false);
      const float nm = norm_s[(ntb + i) * 16 + l15];
      #pragma unroll
      for (int r = 0; r < 8; ++r)
        obase[(size_t)(cT * 16 + r + hi * 8) * N_DIM + n0 + ncol] = acc[r] * nm;
    }
    __syncthreads();
  }
}

extern "C" void kernel_launch(void* const* d_in, const int* in_sizes, int n_in,
                              void* d_out, int out_size, void* d_ws, size_t ws_size,
                              hipStream_t stream) {
  const float* high = (const float*)d_in[0];
  const float* low  = (const float*)d_in[1];
  const float* Wq   = (const float*)d_in[2];
  const float* bq   = (const float*)d_in[3];
  const float* Wk   = (const float*)d_in[4];
  const float* bk   = (const float*)d_in[5];
  const float* Wv   = (const float*)d_in[6];
  const float* bv   = (const float*)d_in[7];
  float* out = (float*)d_out;

  // Workspace budget: B*(P+1)*(4096+64) floats. Choose P from ws_size, cap 128.
  size_t ws_floats = ws_size / sizeof(float);
  long long Pll = (long long)(ws_floats / ((size_t)B_DIM * (C_DIM * C_DIM + C_DIM))) - 1;
  int P = (int)(Pll < 1 ? 1 : (Pll > 128 ? 128 : Pll));

  float* attnP = (float*)d_ws;
  float* ksumP = attnP + (size_t)B_DIM * P * C_DIM * C_DIM;
  float* attnF = ksumP + (size_t)B_DIM * P * C_DIM;
  float* ksumF = attnF + (size_t)B_DIM * C_DIM * C_DIM;

  k1_proj_kv<<<B_DIM * P, 256, 0, stream>>>(low, Wk, bk, Wv, bv, attnP, ksumP, P);
  k2_reduce<<<B_DIM, 256, 0, stream>>>(attnP, ksumP, attnF, ksumF, P);
  k3_out<<<B_DIM * BLK3, 256, 0, stream>>>(high, Wq, bq, attnF, ksumF, out);
}
